// MultiHeadAttentionLayer_29360396436101
// MI455X (gfx1250) — matrix operand order repaired and checked by reading, not compiled, not run on hardware
//
#include <hip/hip_runtime.h>

#define NHEAD 16
#define HDIM  64
#define LSEQ  2048
#define BATCH 2
#define HID   1024

typedef __attribute__((ext_vector_type(16))) _Float16 v16h;
typedef __attribute__((ext_vector_type(8)))  _Float16 v8h;
typedef __attribute__((ext_vector_type(4)))  _Float16 v4h;
typedef __attribute__((ext_vector_type(8)))  float    v8f;
typedef __attribute__((ext_vector_type(4)))  int      v4i;

// ---------------------------------------------------------------------------
// CDNA5 async global->LDS copy (ASYNCcnt path), with portable fallback.
// Builtin signature (from hipcc diagnostic): (AS1 v4i* src, AS3 v4i* dst, imm, imm)
// ---------------------------------------------------------------------------
#if __has_builtin(__builtin_amdgcn_global_load_async_to_lds_b128) && \
    __has_builtin(__builtin_amdgcn_s_wait_asynccnt)
#define USE_ASYNC_LDS 1
#else
#define USE_ASYNC_LDS 0
#endif

#define GLOBAL_AS __attribute__((address_space(1)))
#define LDS_AS    __attribute__((address_space(3)))

__device__ __forceinline__ void cp_b128_g2l(_Float16* l, const _Float16* g) {
#if USE_ASYNC_LDS
  __builtin_amdgcn_global_load_async_to_lds_b128(
      (GLOBAL_AS v4i*)(g), (LDS_AS v4i*)(l), 0, 0);
#else
  *(v8h*)l = *(const v8h*)g;
#endif
}

__device__ __forceinline__ void async_fence() {
#if USE_ASYNC_LDS
  __builtin_amdgcn_s_wait_asynccnt(0);
#endif
}

__device__ __forceinline__ v16h make_frag(v8h lo, v8h hi) {
  v16h r;
#pragma unroll
  for (int i = 0; i < 8; ++i) { r[i] = lo[i]; r[i + 8] = hi[i]; }
  return r;
}

__device__ __forceinline__ v16h frag_from(const _Float16* p0, const _Float16* p1) {
  return make_frag(*(const v8h*)p0, *(const v8h*)p1);
}

// ---------------------------------------------------------------------------
// fp32 -> fp16 converter (4 elements per thread)
// ---------------------------------------------------------------------------
__global__ void cvt_f32_f16(const float* __restrict__ in, _Float16* __restrict__ out, int n4) {
  int i = blockIdx.x * blockDim.x + threadIdx.x;
  if (i >= n4) return;
  float4 f = reinterpret_cast<const float4*>(in)[i];
  v4h h;
  h[0] = (_Float16)f.x; h[1] = (_Float16)f.y; h[2] = (_Float16)f.z; h[3] = (_Float16)f.w;
  reinterpret_cast<v4h*>(out)[i] = h;
}

// ---------------------------------------------------------------------------
// QKV projection: Y[m][n] = sum_k X[m][k] * W[n][k] + bias[n]
// Double-buffered LDS tiles via async global->LDS copies.
// Block: 128(M) x 128(N), BK=32, 256 threads = 8 waves, wave tile 32x64.
// Output fp16 head-split [B][NH][L][HD].
// ---------------------------------------------------------------------------
__global__ __launch_bounds__(256)
void qkv_proj_kernel(const _Float16* __restrict__ X,    // [4096][1024]
                     const _Float16* __restrict__ W,    // [1024][1024], o-major
                     const float*    __restrict__ bias, // [1024]
                     _Float16* __restrict__ Yh)         // [B][NH][L][HD]
{
  __shared__ __align__(16) _Float16 As[2][128 * 32];
  __shared__ __align__(16) _Float16 Bs[2][128 * 32];

  const int tid  = threadIdx.x;
  const int lane = tid & 31;
  const int wave = tid >> 5;
  const int hi   = lane >> 4;
  const int ln   = lane & 15;

  const int blockN = blockIdx.x * 128;
  const int blockM = blockIdx.y * 128;

  const int waveM = (wave & 3) * 32;
  const int waveN = (wave >> 2) * 64;

  v8f acc[2][4] = {};

  const int ldRow = tid >> 1;        // 0..127
  const int ldCol = (tid & 1) * 16;  // 0 or 16
  const _Float16* gA = X + (size_t)(blockM + ldRow) * HID + ldCol;
  const _Float16* gB = W + (size_t)(blockN + ldRow) * HID + ldCol;
  const int sOff = ldRow * 32 + ldCol;

  auto issue = [&](int k0, int buf) {
    cp_b128_g2l(&As[buf][sOff],     gA + k0);
    cp_b128_g2l(&As[buf][sOff + 8], gA + k0 + 8);
    cp_b128_g2l(&Bs[buf][sOff],     gB + k0);
    cp_b128_g2l(&Bs[buf][sOff + 8], gB + k0 + 8);
  };

  issue(0, 0);
  async_fence();
  __syncthreads();

  for (int k0 = 0; k0 < HID; k0 += 32) {
    const int cur = (k0 >> 5) & 1;
    if (k0 + 32 < HID) {
      __builtin_prefetch(gA + k0 + 64, 0, 1);   // global_prefetch_b8, 2 steps ahead
      __builtin_prefetch(gB + k0 + 64, 0, 1);
      issue(k0 + 32, cur ^ 1);                  // async fill of next buffer
    }

    const _Float16* a  = &As[cur][0];
    const _Float16* bb = &Bs[cur][0];

    // load ALL fragments first (independent dests), then back-to-back WMMAs
    v16h afrag[2], bfrag[4];
#pragma unroll
    for (int mt = 0; mt < 2; ++mt) {
      const int row = waveM + mt * 16 + ln;
      afrag[mt] = frag_from(&a[row * 32 + 8 * hi], &a[row * 32 + 16 + 8 * hi]);
    }
#pragma unroll
    for (int nt = 0; nt < 4; ++nt) {
      const int col = waveN + nt * 16 + ln;
      bfrag[nt] = frag_from(&bb[col * 32 + 8 * hi], &bb[col * 32 + 16 + 8 * hi]);
    }
#pragma unroll
    for (int nt = 0; nt < 4; ++nt)
#pragma unroll
      for (int mt = 0; mt < 2; ++mt)
        acc[mt][nt] = __builtin_amdgcn_wmma_f32_16x16x32_f16(
            false, afrag[mt], false, bfrag[nt], (short)0, acc[mt][nt], false, false);

    async_fence();
    __syncthreads();
  }

  // bias + head-split fp16 store
#pragma unroll
  for (int nt = 0; nt < 4; ++nt) {
    const int n  = blockN + waveN + nt * 16 + ln;
    const float bv = bias[n];
    const int nh = n >> 6, hd = n & 63;
#pragma unroll
    for (int mt = 0; mt < 2; ++mt) {
#pragma unroll
      for (int r = 0; r < 8; ++r) {
        const int m  = blockM + waveM + mt * 16 + r + 8 * hi;
        const int bi = m >> 11;       // / 2048
        const int l  = m & 2047;
        Yh[((size_t)((bi * NHEAD + nh) * LSEQ + l)) * HDIM + hd] =
            (_Float16)(acc[mt][nt][r] + bv);
      }
    }
  }
}

// ---------------------------------------------------------------------------
// Flash attention: per (b, h, 256 q rows). 8 waves x 32 q rows each.
// Key blocks of 64; K tile via async global->LDS; online softmax;
// all matmuls via v_wmma_f32_16x16x32_f16.
// ---------------------------------------------------------------------------
__global__ __launch_bounds__(256)
void attention_kernel(const _Float16* __restrict__ Qh,
                      const _Float16* __restrict__ Kh,
                      const _Float16* __restrict__ Vh,
                      const float* __restrict__ amaskG,  // [B][L]
                      const float* __restrict__ hmaskG,  // [NH]
                      float* __restrict__ Out)           // [B][L][H]
{
  __shared__ __align__(16) _Float16 Kt[64 * 64];       // [key][hd]
  __shared__ __align__(16) _Float16 Vt[64 * 64];       // transposed [hd][key]
  __shared__ __align__(16) _Float16 Ps[8][32 * 64];    // per-wave P staging
  __shared__ float Am[64];

  const int tid  = threadIdx.x;
  const int lane = tid & 31;
  const int wave = tid >> 5;
  const int hi   = lane >> 4;
  const int ln   = lane & 15;

  const int h = blockIdx.y;
  const int b = blockIdx.z;
  const int qbase = blockIdx.x * 256 + wave * 32;

  const size_t headOff = (size_t)(b * NHEAD + h) * LSEQ * HDIM;
  const _Float16* Q = Qh + headOff;
  const _Float16* K = Kh + headOff;
  const _Float16* V = Vh + headOff;

  // Q fragments stay resident in registers (32 rows x 64 hd)
  v16h qfrag[2][2];
#pragma unroll
  for (int mt = 0; mt < 2; ++mt)
#pragma unroll
    for (int ks = 0; ks < 2; ++ks) {
      const _Float16* qr = Q + (size_t)(qbase + mt * 16 + ln) * HDIM + ks * 32;
      qfrag[mt][ks] = frag_from(qr + 8 * hi, qr + 16 + 8 * hi);
    }

  v8f oacc[2][4] = {};
  float mrun[2][8], srun[2][8];
#pragma unroll
  for (int mt = 0; mt < 2; ++mt)
#pragma unroll
    for (int r = 0; r < 8; ++r) { mrun[mt][r] = -3.0e38f; srun[mt][r] = 0.0f; }

  const int ldKey = tid >> 2;        // 0..63
  const int ldHd  = (tid & 3) * 16;  // 0,16,32,48
  const float scale = 0.125f;        // 1/sqrt(64)

  _Float16* pw = &Ps[wave][0];

  for (int kb = 0; kb < LSEQ; kb += 64) {
    __syncthreads();  // all waves done with previous K/V tiles
    {
      const _Float16* kr = K + (size_t)(kb + ldKey) * HDIM + ldHd;
      const _Float16* vr = V + (size_t)(kb + ldKey) * HDIM + ldHd;
      // K tile: straight async DMA into LDS (same layout)
      cp_b128_g2l(&Kt[ldKey * 64 + ldHd],     kr);
      cp_b128_g2l(&Kt[ldKey * 64 + ldHd + 8], kr + 8);
      // V tile: needs transpose -> manual
      v8h vv0 = *(const v8h*)(vr);
      v8h vv1 = *(const v8h*)(vr + 8);
#pragma unroll
      for (int j = 0; j < 8; ++j) {
        Vt[(ldHd + j) * 64 + ldKey]     = vv0[j];
        Vt[(ldHd + 8 + j) * 64 + ldKey] = vv1[j];
      }
      if (tid < 64) Am[tid] = amaskG[(size_t)b * LSEQ + kb + tid];
    }
    async_fence();
    __syncthreads();

    // ---- S = scale * Q K^T + mask ; online softmax ; P -> LDS (fp16) ----
#pragma unroll
    for (int mt = 0; mt < 2; ++mt) {
      v8f s[4];
#pragma unroll
      for (int nt = 0; nt < 4; ++nt) s[nt] = (v8f){};
#pragma unroll
      for (int ks = 0; ks < 2; ++ks) {
        v16h bfrag[4];
#pragma unroll
        for (int nt = 0; nt < 4; ++nt) {
          const int key = nt * 16 + ln;
          bfrag[nt] = frag_from(&Kt[key * 64 + ks * 32 + 8 * hi],
                                &Kt[key * 64 + ks * 32 + 16 + 8 * hi]);
        }
#pragma unroll
        for (int nt = 0; nt < 4; ++nt)
          s[nt] = __builtin_amdgcn_wmma_f32_16x16x32_f16(
              false, qfrag[mt][ks], false, bfrag[nt], (short)0, s[nt], false, false);
      }
#pragma unroll
      for (int nt = 0; nt < 4; ++nt) {
        const float am = Am[nt * 16 + ln];
#pragma unroll
        for (int r = 0; r < 8; ++r) s[nt][r] = s[nt][r] * scale + am;
      }
      // block row-max across n-tiles then across the 16-lane half
#pragma unroll
      for (int r = 0; r < 8; ++r) {
        float c = s[0][r];
#pragma unroll
        for (int nt = 1; nt < 4; ++nt) c = fmaxf(c, s[nt][r]);
        c = fmaxf(c, __shfl_xor(c, 1));
        c = fmaxf(c, __shfl_xor(c, 2));
        c = fmaxf(c, __shfl_xor(c, 4));
        c = fmaxf(c, __shfl_xor(c, 8));

        const float mnew  = fmaxf(mrun[mt][r], c);
        const float alpha = __expf(mrun[mt][r] - mnew);
        mrun[mt][r] = mnew;

        float psum = 0.0f;
#pragma unroll
        for (int nt = 0; nt < 4; ++nt) {
          const float p = __expf(s[nt][r] - mnew);
          s[nt][r] = p;
          psum += p;
        }
        psum += __shfl_xor(psum, 1);
        psum += __shfl_xor(psum, 2);
        psum += __shfl_xor(psum, 4);
        psum += __shfl_xor(psum, 8);
        srun[mt][r] = srun[mt][r] * alpha + psum;
#pragma unroll
        for (int nt = 0; nt < 4; ++nt) oacc[mt][nt][r] *= alpha;
      }
      // stage P as fp16 in this wave's LDS region
#pragma unroll
      for (int nt = 0; nt < 4; ++nt)
#pragma unroll
        for (int r = 0; r < 8; ++r)
          pw[(mt * 16 + r + 8 * hi) * 64 + nt * 16 + ln] = (_Float16)s[nt][r];
    }

    // ---- O += P @ V : per ks, load 2 A-frags + 4 B-frags, then 8 WMMAs ----
#pragma unroll
    for (int ks = 0; ks < 2; ++ks) {
      v16h afrag[2], bfrag[4];
#pragma unroll
      for (int mt = 0; mt < 2; ++mt) {
        const _Float16* pr = &pw[(mt * 16 + ln) * 64 + ks * 32];
        afrag[mt] = frag_from(pr + 8 * hi, pr + 16 + 8 * hi);
      }
#pragma unroll
      for (int nt = 0; nt < 4; ++nt) {
        const int hd = nt * 16 + ln;
        bfrag[nt] = frag_from(&Vt[hd * 64 + ks * 32 + 8 * hi],
                              &Vt[hd * 64 + ks * 32 + 16 + 8 * hi]);
      }
#pragma unroll
      for (int nt = 0; nt < 4; ++nt)
#pragma unroll
        for (int mt = 0; mt < 2; ++mt)
          oacc[mt][nt] = __builtin_amdgcn_wmma_f32_16x16x32_f16(
              false, afrag[mt], false, bfrag[nt], (short)0, oacc[mt][nt], false, false);
    }
  }

  // normalize, head_mask, store fp32 [B][L][H]
  const float hm = hmaskG[h];
#pragma unroll
  for (int mt = 0; mt < 2; ++mt)
#pragma unroll
    for (int nt = 0; nt < 4; ++nt) {
      const int hd = nt * 16 + ln;
#pragma unroll
      for (int r = 0; r < 8; ++r) {
        const int l = qbase + mt * 16 + r + 8 * hi;
        Out[((size_t)b * LSEQ + l) * HID + h * HDIM + hd] =
            oacc[mt][nt][r] / srun[mt][r] * hm;
      }
    }
}

// ---------------------------------------------------------------------------
extern "C" void kernel_launch(void* const* d_in, const int* in_sizes, int n_in,
                              void* d_out, int out_size, void* d_ws, size_t ws_size,
                              hipStream_t stream) {
  const float* query = (const float*)d_in[0];
  const float* key   = (const float*)d_in[1];
  const float* value = (const float*)d_in[2];
  const float* amask = (const float*)d_in[3];
  const float* hmask = (const float*)d_in[4];
  const float* Wq    = (const float*)d_in[5];
  const float* bq    = (const float*)d_in[6];
  const float* Wk    = (const float*)d_in[7];
  const float* bk    = (const float*)d_in[8];
  const float* Wv    = (const float*)d_in[9];
  const float* bv    = (const float*)d_in[10];
  float* out = (float*)d_out;

  const size_t NX = (size_t)BATCH * LSEQ * HID;  // 4,194,304
  const size_t NW = (size_t)HID * HID;           // 1,048,576

  _Float16* ws = (_Float16*)d_ws;
  _Float16* q16  = ws;                 // fp16 activations
  _Float16* k16  = q16 + NX;
  _Float16* v16  = k16 + NX;
  _Float16* wq16 = v16 + NX;           // fp16 weights [o][h]
  _Float16* wk16 = wq16 + NW;
  _Float16* wv16 = wk16 + NW;
  _Float16* Qh   = wv16 + NW;          // head-split [B][NH][L][HD]
  _Float16* Kh   = Qh + NX;
  _Float16* Vh   = Kh + NX;

  // 1) fp32 -> fp16 conversion
  {
    const int tx = 256;
    int n4 = (int)(NX / 4);
    cvt_f32_f16<<<(n4 + tx - 1) / tx, tx, 0, stream>>>(query, q16, n4);
    cvt_f32_f16<<<(n4 + tx - 1) / tx, tx, 0, stream>>>(key,   k16, n4);
    cvt_f32_f16<<<(n4 + tx - 1) / tx, tx, 0, stream>>>(value, v16, n4);
    int w4 = (int)(NW / 4);
    cvt_f32_f16<<<(w4 + tx - 1) / tx, tx, 0, stream>>>(Wq, wq16, w4);
    cvt_f32_f16<<<(w4 + tx - 1) / tx, tx, 0, stream>>>(Wk, wk16, w4);
    cvt_f32_f16<<<(w4 + tx - 1) / tx, tx, 0, stream>>>(Wv, wv16, w4);
  }

  // 2) QKV projections (WMMA GEMMs), head-split fp16 outputs
  {
    dim3 grid(HID / 128, (BATCH * LSEQ) / 128);  // 8 x 32
    qkv_proj_kernel<<<grid, 256, 0, stream>>>(q16, wq16, bq, Qh);
    qkv_proj_kernel<<<grid, 256, 0, stream>>>(k16, wk16, bk, Kh);
    qkv_proj_kernel<<<grid, 256, 0, stream>>>(v16, wv16, bv, Vh);
  }

  // 3) flash attention (WMMA), fp32 output
  {
    dim3 grid(LSEQ / 256, NHEAD, BATCH);  // 8 x 16 x 2
    attention_kernel<<<grid, 256, 0, stream>>>(Qh, Kh, Vh, amask, hmask, out);
  }
}
